// ParametricSelfAttention_32220844655349
// MI455X (gfx1250) — matrix operand order repaired and checked by reading, not compiled, not run on hardware
//
#include <hip/hip_runtime.h>

typedef __attribute__((ext_vector_type(16))) _Float16     v16h;
typedef __attribute__((ext_vector_type(8)))  float        v8f;
typedef __attribute__((ext_vector_type(4)))  unsigned int u32x4;
typedef __attribute__((ext_vector_type(4)))  int          i32x4;
typedef __attribute__((ext_vector_type(8)))  int          i32x8;

namespace {

constexpr int kB = 8;
constexpr int kS = 2048;
constexpr int kD = 512;

__device__ __forceinline__ v8f wmma16(v16h a, v16h b, v8f c) {
  // D = A(16x32,f16) * B(32x16,f16) + C(16x16,f32)
  return __builtin_amdgcn_wmma_f32_16x16x32_f16(
      /*neg_a=*/false, a, /*neg_b=*/false, b,
      /*c_mod=*/(short)0, c, /*reuse_a=*/false, /*reuse_b=*/false);
}

// ---- fragment loaders (contiguous k order, same for A and B) ----------------------------
// A (16x32 MxK): m = lane&15, half = lane>>4, k(i,j) = 16*half+2*i+j
// B (32x16 KxN): n = lane&15, half = lane>>4, k(i,j) = 16*half+2*i+j

__device__ __forceinline__ v16h load_a_f32(const float* src, int row0, int k0, int ld) {
  const int lane = threadIdx.x & 31;
  const int m = lane & 15, half = lane >> 4;
  const float* p = src + (size_t)(row0 + m) * ld + k0;
  v16h a;
#pragma unroll
  for (int i = 0; i < 8; ++i) {
    const int kb = 16 * half + 2 * i;
    a[2 * i]     = (_Float16)p[kb];
    a[2 * i + 1] = (_Float16)p[kb + 1];
  }
  return a;
}

__device__ __forceinline__ v16h load_a_f16(const _Float16* src, int row0, int k0, int ld) {
  const int lane = threadIdx.x & 31;
  const int m = lane & 15, half = lane >> 4;
  const _Float16* p = src + (size_t)(row0 + m) * ld + k0;
  v16h a;
#pragma unroll
  for (int i = 0; i < 8; ++i) {
    const int kb = 16 * half + 2 * i;
    a[2 * i]     = p[kb];
    a[2 * i + 1] = p[kb + 1];
  }
  return a;
}

// B[k][n] from row-major src[k][n]
__device__ __forceinline__ v16h load_b_rm_f16(const _Float16* src, int k0, int n0, int ld) {
  const int lane = threadIdx.x & 31;
  const int n = lane & 15, half = lane >> 4;
  const _Float16* p = src + (size_t)(k0 + 16 * half) * ld + n0 + n;
  v16h b;
#pragma unroll
  for (int i = 0; i < 8; ++i) {
    b[2 * i]     = p[(size_t)(2 * i) * ld];
    b[2 * i + 1] = p[(size_t)(2 * i + 1) * ld];
  }
  return b;
}

// B[k][n] from row-major src[n][k] (B = src^T) -> 16 contiguous halfs per lane
__device__ __forceinline__ v16h load_b_tr_f16(const _Float16* src, int k0, int n0, int ld) {
  const int lane = threadIdx.x & 31;
  const int n = lane & 15, half = lane >> 4;
  const _Float16* p = src + (size_t)(n0 + n) * ld + k0 + 16 * half;
  v16h b;
#pragma unroll
  for (int i = 0; i < 16; ++i) b[i] = p[i];
  return b;
}

__device__ __forceinline__ v16h load_b_tr_f32(const float* src, int k0, int n0, int ld) {
  const int lane = threadIdx.x & 31;
  const int n = lane & 15, half = lane >> 4;
  const float* p = src + (size_t)(n0 + n) * ld + k0 + 16 * half;
  v16h b;
#pragma unroll
  for (int i = 0; i < 16; ++i) b[i] = (_Float16)p[i];
  return b;
}

// ---- kernel 1: c = context @ W_in^T, stored f16 -----------------------------
__global__ __launch_bounds__(32)
void pattn_gemm_in(const float* __restrict__ X,      // [kB*kS, kD]
                   const float* __restrict__ W,      // [kD, kD]
                   _Float16* __restrict__ C16) {     // [kB*kS, kD]
  const int n0 = blockIdx.x * 16;
  const int m0 = blockIdx.y * 16;
  v8f acc = {};
#pragma unroll 4
  for (int k0 = 0; k0 < kD; k0 += 32) {
    __builtin_prefetch(X + (size_t)m0 * kD + k0 + 64, 0, 0);
    v16h a = load_a_f32(X, m0, k0, kD);
    v16h b = load_b_tr_f32(W, k0, n0, kD);   // B[k][n] = W[n][k]
    acc = wmma16(a, b, acc);
  }
  const int lane = threadIdx.x & 31;
  const int n = lane & 15, half = lane >> 4;
#pragma unroll
  for (int j = 0; j < 8; ++j)
    C16[(size_t)(m0 + j + 8 * half) * kD + n0 + n] = (_Float16)acc[j];
}

// ---- kernel 2: fused scores -> softmax -> attn -> ctx -> tanh ---------------
// 2 waves / WG, 32 query rows. Dynamic LDS (CDNA5 320KB WGP):
//   sc     : 32*kS*4  = 256 KB  (score block, f32)
//   astage : 32*kD*2  =  32 KB  (A rows, staged by TDM tensor_load_to_lds)
//   bstage : 16*kD*2  =  16 KB  (B tile, staged by global_load_async_to_lds)
__global__ __launch_bounds__(64)
void pattn_attn(const _Float16* __restrict__ C16,    // [kB, kS, kD] f16
                float* __restrict__ attn,            // [kB, kS, kS] f32 out
                _Float16* __restrict__ H16) {        // [kB, kS, kD] tanh(ctx) f16
  extern __shared__ char dynlds[];
  float*    sc     = (float*)dynlds;
  _Float16* astage = (_Float16*)(dynlds + (size_t)32 * kS * 4);
  _Float16* bstage = (_Float16*)(dynlds + (size_t)32 * kS * 4 + (size_t)32 * kD * 2);
  __shared__ float rmax[32], rinv[32];

  const int b  = blockIdx.y;
  const int s0 = blockIdx.x * 32;
  const _Float16* cb = C16 + (size_t)b * kS * kD;
  const int wave = threadIdx.x >> 5;
  const int lane = threadIdx.x & 31;
  const int n = lane & 15, half = lane >> 4;

  const unsigned dynbase    = __builtin_amdgcn_groupstaticsize();
  const unsigned astage_lds = dynbase + 32u * kS * 4u;
  const unsigned bstage_lds = astage_lds + 32u * kD * 2u;

  // --- TDM: stage A block c[s0..s0+32)[0..kD) into astage (one wave issues) --
  if (threadIdx.x < 32) {
    const unsigned long long ga =
        (unsigned long long)(uintptr_t)(cb + (size_t)s0 * kD);
    u32x4 g0;
    g0[0] = 1u;                                             // count=1, user mode
    g0[1] = astage_lds;                                     // lds_addr
    g0[2] = (unsigned)(ga & 0xFFFFFFFFu);                   // global_addr lo
    g0[3] = (unsigned)((ga >> 32) & 0x1FFFFFFu) | (2u << 30); // addr hi | type=2
    i32x8 g1;
    g1[0] = (int)(1u << 16);                                // data_size=2B, mask=0
    g1[1] = (int)((unsigned)kD << 16);                      // tensor_dim0[15:0]
    g1[2] = (int)(((unsigned)kD >> 16) | ((unsigned)kS << 16)); // dim0 hi | dim1 lo
    g1[3] = (int)(((unsigned)kS >> 16) | ((unsigned)kD << 16)); // dim1 hi | tile_dim0
    g1[4] = 32;                                             // tile_dim1=32, tile_dim2=0
    g1[5] = kD;                                             // tensor_dim0_stride lo32
    g1[6] = 0;
    g1[7] = 0;
    i32x4 z4 = {0, 0, 0, 0};
#if defined(__clang_major__) && (__clang_major__ >= 23)
    i32x8 z8 = {0, 0, 0, 0, 0, 0, 0, 0};
    __builtin_amdgcn_tensor_load_to_lds(g0, g1, z4, z4, z8, 0);
#else
    __builtin_amdgcn_tensor_load_to_lds(g0, g1, z4, z4, 0);
#endif
    __builtin_amdgcn_s_wait_tensorcnt(0);
  }
  __syncthreads();

  // --- hoist all 16 A fragments for this wave's 16 query rows ---------------
  v16h afr[16];
#pragma unroll
  for (int kk = 0; kk < 16; ++kk)
    afr[kk] = load_a_f16(astage, wave * 16, kk * 32, kD);

  // --- scores: 128 column tiles, B staged cooperatively via async->LDS ------
  for (int t0 = 0; t0 < kS; t0 += 16) {
    // stage c rows [t0..t0+16) x kD (16 KB): 1024 x 16B chunks over 64 threads
    for (int ch = threadIdx.x; ch < 1024; ch += 64) {
      const unsigned ldsb = bstage_lds + (unsigned)ch * 16u;
      const char* g = (const char*)(cb + (size_t)(t0 + (ch >> 6)) * kD) + (ch & 63) * 16;
      asm volatile("global_load_async_to_lds_b128 %0, %1, off"
                   :: "v"(ldsb), "v"(g) : "memory");
    }
    asm volatile("s_wait_asynccnt 0x0" ::: "memory");
    __syncthreads();

    v8f acc = {};
#pragma unroll
    for (int kk = 0; kk < 16; ++kk) {
      v16h bt = load_b_tr_f16(bstage, kk * 32, 0, kD);  // B = c^T slice (from LDS)
      acc = wmma16(afr[kk], bt, acc);
    }
#pragma unroll
    for (int j = 0; j < 8; ++j)
      sc[(size_t)(wave * 16 + j + 8 * half) * kS + t0 + n] = acc[j];
    __syncthreads();   // protect bstage before next tile's staging
  }
  __syncthreads();

  // --- per-row max and 1/sum(exp) -------------------------------------------
  if (threadIdx.x < 32) {
    const float* r = sc + (size_t)threadIdx.x * kS;
    float m = -3.4e38f;
    for (int t = 0; t < kS; ++t) m = fmaxf(m, r[t]);
    float s = 0.f;
    for (int t = 0; t < kS; ++t) s += __expf(r[t] - m);
    rmax[threadIdx.x] = m;
    rinv[threadIdx.x] = 1.0f / s;
  }
  __syncthreads();

  // --- normalize in LDS, stream attn rows to HBM (contiguous) ---------------
  float* attn_blk = attn + ((size_t)b * kS + s0) * kS;
  for (int idx = threadIdx.x; idx < 32 * kS; idx += 64) {
    const int r = idx >> 11;                               // kS == 2048
    const float v = __expf(sc[idx] - rmax[r]) * rinv[r];
    sc[idx] = v;
    attn_blk[idx] = v;
  }
  __syncthreads();

  // --- ctx = P @ c ; H = tanh(ctx) f16. 4-tile N-chunks reuse each A frag ---
  const float* pw = sc + (size_t)wave * 16 * kS;
  for (int nc = 0; nc < kD; nc += 64) {
    v8f acc[4] = {};
    for (int k0 = 0; k0 < kS; k0 += 32) {
      v16h a = load_a_f32(pw, 0, k0, kS);                  // from LDS, cvt f16
#pragma unroll
      for (int t = 0; t < 4; ++t) {
        v16h bm = load_b_rm_f16(cb, k0, nc + t * 16, kD);  // B[k][n] = c[k][n]
        acc[t] = wmma16(a, bm, acc[t]);
      }
    }
#pragma unroll
    for (int t = 0; t < 4; ++t) {
#pragma unroll
      for (int j = 0; j < 8; ++j) {
        const int m = s0 + wave * 16 + j + 8 * half;
        H16[((size_t)b * kS + m) * kD + nc + t * 16 + n] = (_Float16)tanhf(acc[t][j]);
      }
    }
  }
}

// ---- kernel 3: out = H @ W_out^T, f32 out -----------------------------------
__global__ __launch_bounds__(32)
void pattn_gemm_out(const _Float16* __restrict__ H,  // [kB*kS, kD] f16
                    const float* __restrict__ W,     // [kD, kD]
                    float* __restrict__ Out) {       // [kB*kS, kD] f32
  const int n0 = blockIdx.x * 16;
  const int m0 = blockIdx.y * 16;
  v8f acc = {};
#pragma unroll 4
  for (int k0 = 0; k0 < kD; k0 += 32) {
    v16h a = load_a_f16(H, m0, k0, kD);
    v16h b = load_b_tr_f32(W, k0, n0, kD);
    acc = wmma16(a, b, acc);
  }
  const int lane = threadIdx.x & 31;
  const int n = lane & 15, half = lane >> 4;
#pragma unroll
  for (int j = 0; j < 8; ++j)
    Out[(size_t)(m0 + j + 8 * half) * kD + n0 + n] = acc[j];
}

} // anonymous namespace

extern "C" void kernel_launch(void* const* d_in, const int* in_sizes, int n_in,
                              void* d_out, int out_size, void* d_ws, size_t ws_size,
                              hipStream_t stream) {
  (void)in_sizes; (void)n_in; (void)out_size; (void)ws_size;

  const float* context = (const float*)d_in[0];   // [kB,kS,kD]
  const float* W_in    = (const float*)d_in[1];   // [kD,kD]
  const float* W_out   = (const float*)d_in[2];   // [kD,kD]

  float* out  = (float*)d_out;                    // [kB,kS,kD]
  float* attn = out + (size_t)kB * kS * kD;       // [kB,kS,kS]

  _Float16* c16 = (_Float16*)d_ws;                // [kB,kS,kD]
  _Float16* h16 = c16 + (size_t)kB * kS * kD;     // [kB,kS,kD]

  // kernel 1: c = context @ W_in^T (f16 result)
  {
    dim3 grid(kD / 16, (kB * kS) / 16);
    pattn_gemm_in<<<grid, 32, 0, stream>>>(context, W_in, c16);
  }
  // kernel 2: fused scores/softmax/attn/ctx/tanh
  {
    dim3 grid(kS / 32, kB);
    const size_t lds_bytes = (size_t)32 * kS * 4       // sc
                           + (size_t)32 * kD * 2       // astage
                           + (size_t)16 * kD * 2;      // bstage  (= 304 KB)
    pattn_attn<<<grid, 64, lds_bytes, stream>>>(c16, attn, h16);
  }
  // kernel 3: out = tanh(ctx) @ W_out^T
  {
    dim3 grid(kD / 16, (kB * kS) / 16);
    pattn_gemm_out<<<grid, 32, 0, stream>>>(h16, W_out, out);
  }
}
